// EmbeddingBlock_13005160972650
// MI455X (gfx1250) — matrix operand order repaired and checked by reading, not compiled, not run on hardware
//
#include <hip/hip_runtime.h>
#include <hip/hip_bf16.h>

// ---------------------------------------------------------------------------
// SchNet-style embedding block for MI455X (gfx1250, wave32, WMMA).
// GEMM orientation: A = weights^T (M = feature cols), B = per-edge/per-atom
// activations (N = edge/atom). D-tile then keeps each lane's 8 results on its
// OWN edge/atom at 8 consecutive feature columns -> no cross-lane shuffles,
// contiguous atomics/stores, vectorized LDS/global accesses.
//   rbf_feats = W_rbf^T @ (env*rbf)                -> bf16 WMMA, f32 accum
//   edge      = rbf_feats * atom_embed[z[nbr1]]
//   aggr      = segment_sum(edge, nbr0)            -> global_atomic_add_f32 (L2)
//   out       = W_cat^T @ [node_emb | aggr] + b    -> bf16 WMMA, f32 accum
// ---------------------------------------------------------------------------

typedef __attribute__((ext_vector_type(16))) __bf16 v16bf;
typedef __attribute__((ext_vector_type(8)))  float  v8f;

#define DIM     128
#define EMBPAD  132   // LDS row stride for atom_embed (132 % 64 = 4 -> de-bank)
#define NRBF    20
#define CUTOFF  5.0f
#define GAMMA   10.0f

// A-matrix (16x32 bf16) K index for element j of this lane's v16bf.
__device__ __forceinline__ int kmapA(int j, bool hi) {
    return j + (j >= 8 ? 8 : 0) + (hi ? 8 : 0);
}

// ---------------------------------------------------------------------------
// Kernel 1: edges. One wave per 16-edge tile (grid-stride).
// LDS: atom_embed padded to 132 f32/row (52800B) + W_rbf A-ready bf16 (8KB).
// ---------------------------------------------------------------------------
__global__ __launch_bounds__(256)
void edge_kernel(const int* __restrict__ z_number,
                 const int* __restrict__ nbrs,
                 const float* __restrict__ dist,
                 const float* __restrict__ atom_embed,
                 const float* __restrict__ W_rbf,
                 float* aggr, int E)
{
    extern __shared__ char smem[];
    float*  sEmb = reinterpret_cast<float*>(smem);                     // 100*132 f32
    __bf16* sA   = reinterpret_cast<__bf16*>(smem + 100 * EMBPAD * 4); // 8*32*16 bf16

    const int tid  = threadIdx.x;
    const int lane = tid & 31;
    const int wv   = tid >> 5;
    const int wavesPerBlock = blockDim.x >> 5;
    const bool hi  = lane >= 16;
    const int l15  = lane & 15;

    // Stage atom_embed into LDS with padded row stride (bank-conflict-free
    // column-range gathers across random rows).
    for (int i = tid; i < 100 * DIM; i += blockDim.x) {
        const int rr = i / DIM, cc = i - rr * DIM;
        sEmb[rr * EMBPAD + cc] = atom_embed[i];
    }

    // Stage W_rbf^T in WMMA-A-ready bf16 layout, K padded 20->32 with zeros.
    // sA[((t*32 + lane)*16) + j] = W_rbf[j + (hi ? 16 : 0)][16t + (lane&15)]
    for (int i = tid; i < 8 * 32 * 16; i += blockDim.x) {
        int j  = i & 15;
        int ln = (i >> 4) & 31;
        int t  = i >> 9;
        int k  = j + ((ln & 16) ? 16 : 0);
        int c  = t * 16 + (ln & 15);
        sA[i] = (k < NRBF) ? (__bf16)W_rbf[k * DIM + c] : (__bf16)0.0f;
    }
    __syncthreads();

    // Hoist the 8 constant A-tiles (W_rbf^T) into VGPRs for the whole loop.
    v16bf Areg[8];
    #pragma unroll
    for (int t = 0; t < 8; ++t)
        Areg[t] = *reinterpret_cast<const v16bf*>(&sA[(t * 32 + lane) * 16]);

    const int numTiles = E >> 4;                 // E is a multiple of 16
    const int stride   = gridDim.x * wavesPerBlock;
    const float mustep = CUTOFF / (float)(NRBF - 1);
    const float kpi    = 3.14159265358979323846f / CUTOFF;
    const int kbase    = hi ? 16 : 0;            // this lane's B K-range
    const int coff     = hi ? 8 : 0;             // this lane's D column sub-offset

    for (int tile = blockIdx.x * wavesPerBlock + wv; tile < numTiles; tile += stride) {
        // Lanes L and L+16 both own edge slot (L&15) of this tile.
        const int e    = tile * 16 + l15;
        const float d  = dist[e];
        const int dst  = nbrs[2 * e + 0];
        const int src  = nbrs[2 * e + 1];

        // Prefetch next tile's edge data (global_prefetch_b8; non-binding).
        if (tile + stride < numTiles) {
            const int en = e + stride * 16;
            __builtin_prefetch(&dist[en], 0, 1);
            __builtin_prefetch(&nbrs[2 * en], 0, 1);
        }

        const float env = (d < CUTOFF) ? 0.5f * (__cosf(kpi * d) + 1.0f) : 0.0f;

        // Build B (32x16 bf16): column n = this lane's edge, K = kbase..kbase+15.
        // env folded in (per-column scalar commutes with the GEMM).
        v16bf B;
        #pragma unroll
        for (int j = 0; j < 16; ++j) {
            const int k = kbase + j;
            float v = 0.0f;
            if (k < NRBF) {
                const float dm = d - (float)k * mustep;
                v = env * __expf(-GAMMA * dm * dm);
            }
            B[j] = (__bf16)v;
        }

        // Per-lane base pointers: own edge's nbr-emb row and aggr row.
        const float* embP = sEmb + z_number[src] * EMBPAD + coff;
        float*       aggP = aggr + dst * DIM + coff;

        // 8 column tiles: 1 WMMA each; lane's 8 results are 8 consecutive
        // feature columns of its own edge -> vector LDS read + offset atomics.
        #pragma unroll
        for (int t = 0; t < 8; ++t) {
            v8f acc = {};
            acc = __builtin_amdgcn_wmma_f32_16x16x32_bf16(
                false, Areg[t], false, B, (short)0, acc, false, false);
            const int c0 = t * 16;
            const float4 e0 = *reinterpret_cast<const float4*>(embP + c0);
            const float4 e1 = *reinterpret_cast<const float4*>(embP + c0 + 4);
            atomicAdd(&aggP[c0 + 0], acc[0] * e0.x);
            atomicAdd(&aggP[c0 + 1], acc[1] * e0.y);
            atomicAdd(&aggP[c0 + 2], acc[2] * e0.z);
            atomicAdd(&aggP[c0 + 3], acc[3] * e0.w);
            atomicAdd(&aggP[c0 + 4], acc[4] * e1.x);
            atomicAdd(&aggP[c0 + 5], acc[5] * e1.y);
            atomicAdd(&aggP[c0 + 6], acc[6] * e1.z);
            atomicAdd(&aggP[c0 + 7], acc[7] * e1.w);
        }
    }
}

// ---------------------------------------------------------------------------
// Kernel 2: nodes. out = W_cat^T @ [node_emb | aggr] + b_cat (per atom).
// One wave per 16-atom tile; K=256 in 8 steps of 32; 8 column-tile f32
// accumulators in VGPRs. W_cat^T staged in 64KB dynamic LDS (bf16, A-ready).
// B-build: each lane's K-slice is ONE contiguous 16-float run of its own
// atom's [node_emb|aggr] row (never straddles the boundary) -> 4x b128 loads.
// Stores: 8 consecutive floats of the lane's own output row -> 2x b128 per t.
// NOTE: `aggr` and `out` alias (same buffer); each wave finishes all aggr
// reads for its rows before storing those rows, and rows are wave-private.
// ---------------------------------------------------------------------------
__global__ __launch_bounds__(256)
void node_kernel(const int* __restrict__ z_number,
                 const float* __restrict__ atom_embed,
                 const float* __restrict__ W_cat,
                 const float* __restrict__ b_cat,
                 const float* aggr, float* out, int N)
{
    extern __shared__ __bf16 sW[]; // 8 ksteps * 8 tiles * 32 lanes * 16 = 32768 bf16

    const int tid  = threadIdx.x;
    const int lane = tid & 31;
    const int wv   = tid >> 5;
    const int wavesPerBlock = blockDim.x >> 5;
    const bool hi  = lane >= 16;
    const int l15  = lane & 15;
    const int coff = hi ? 8 : 0;

    // Stage W_cat^T (256x128 f32) into bf16 A-ready layout.
    // sW[(((ks*8 + t)*32 + lane)*16) + j] = W_cat[32ks + j + (hi ? 16 : 0)][16t + (lane&15)]
    for (int i = tid; i < 8 * 8 * 32 * 16; i += blockDim.x) {
        int j  = i & 15;
        int ln = (i >> 4) & 31;
        int t  = (i >> 9) & 7;
        int ks = i >> 12;
        int k  = 32 * ks + j + ((ln & 16) ? 16 : 0);
        int c  = 16 * t + (ln & 15);
        sW[i] = (__bf16)W_cat[k * DIM + c];
    }
    __syncthreads();

    const int numTiles = N >> 4;                 // N is a multiple of 16
    const int stride   = gridDim.x * wavesPerBlock;

    for (int tile = blockIdx.x * wavesPerBlock + wv; tile < numTiles; tile += stride) {
        const int atom = tile * 16 + l15;        // this lane's B-column (N = lane&15)
        const float* __restrict__ embRow = atom_embed + z_number[atom] * DIM;
        const float* aggRow = aggr + (size_t)atom * DIM;

        v8f acc[8] = {};                         // 8 column tiles of 16x16 f32

        #pragma unroll
        for (int ks = 0; ks < 8; ++ks) {
            // This lane's B K-slice: one contiguous run of 16.
            const int base = 32 * ks + (hi ? 16 : 0);
            const float* p = (base < DIM) ? (embRow + base) : (aggRow + base - DIM);
            const float4 q0 = *reinterpret_cast<const float4*>(p + 0);
            const float4 q1 = *reinterpret_cast<const float4*>(p + 4);
            const float4 q2 = *reinterpret_cast<const float4*>(p + 8);
            const float4 q3 = *reinterpret_cast<const float4*>(p + 12);

            v16bf B;
            B[0]  = (__bf16)q0.x; B[1]  = (__bf16)q0.y;
            B[2]  = (__bf16)q0.z; B[3]  = (__bf16)q0.w;
            B[4]  = (__bf16)q1.x; B[5]  = (__bf16)q1.y;
            B[6]  = (__bf16)q1.z; B[7]  = (__bf16)q1.w;
            B[8]  = (__bf16)q2.x; B[9]  = (__bf16)q2.y;
            B[10] = (__bf16)q2.z; B[11] = (__bf16)q2.w;
            B[12] = (__bf16)q3.x; B[13] = (__bf16)q3.y;
            B[14] = (__bf16)q3.z; B[15] = (__bf16)q3.w;

            #pragma unroll
            for (int t = 0; t < 8; ++t) {
                const v16bf A = *reinterpret_cast<const v16bf*>(
                    &sW[((ks * 8 + t) * 32 + lane) * 16]);
                acc[t] = __builtin_amdgcn_wmma_f32_16x16x32_bf16(
                    false, A, false, B, (short)0, acc[t], false, false);
            }
        }

        // Store with bias: lane's own atom row, 8 consecutive cols per tile t.
        float* outP = out + (size_t)atom * DIM + coff;
        #pragma unroll
        for (int t = 0; t < 8; ++t) {
            const int c0 = t * 16;
            const float4 b0 = *reinterpret_cast<const float4*>(b_cat + c0 + coff);
            const float4 b1 = *reinterpret_cast<const float4*>(b_cat + c0 + coff + 4);
            float4 o0, o1;
            o0.x = acc[t][0] + b0.x; o0.y = acc[t][1] + b0.y;
            o0.z = acc[t][2] + b0.z; o0.w = acc[t][3] + b0.w;
            o1.x = acc[t][4] + b1.x; o1.y = acc[t][5] + b1.y;
            o1.z = acc[t][6] + b1.z; o1.w = acc[t][7] + b1.w;
            *reinterpret_cast<float4*>(outP + c0)     = o0;
            *reinterpret_cast<float4*>(outP + c0 + 4) = o1;
        }
    }
}

// ---------------------------------------------------------------------------
extern "C" void kernel_launch(void* const* d_in, const int* in_sizes, int n_in,
                              void* d_out, int out_size, void* d_ws, size_t ws_size,
                              hipStream_t stream) {
    const int*   z_number   = (const int*)  d_in[0];
    const int*   nbrs       = (const int*)  d_in[1];
    const float* dist       = (const float*)d_in[2];
    const float* atom_embed = (const float*)d_in[3];
    const float* W_rbf      = (const float*)d_in[4];
    const float* W_cat      = (const float*)d_in[5];
    const float* b_cat      = (const float*)d_in[6];

    const int N = in_sizes[0];       // 50000
    const int E = in_sizes[2];       // 1600000

    float* out  = (float*)d_out;
    float* aggr = out;               // reuse output buffer as segment-sum target

    (void)d_ws; (void)ws_size; (void)n_in; (void)out_size;

    // Zero the aggregation buffer (capturable).
    hipMemsetAsync(aggr, 0, (size_t)N * DIM * sizeof(float), stream);

    // Edge pass: 1024 blocks x 8 waves, grid-stride over E/16 = 100000 tiles.
    const size_t lds1 = (size_t)100 * EMBPAD * 4 + (size_t)8 * 32 * 16 * 2; // 60992 B
    edge_kernel<<<1024, 256, lds1, stream>>>(z_number, nbrs, dist,
                                             atom_embed, W_rbf, aggr, E);

    // Node pass: 200 blocks x 8 waves, grid-stride over N/16 = 3125 tiles.
    const size_t lds2 = (size_t)8 * 8 * 32 * 16 * 2;                        // 65536 B
    node_kernel<<<200, 256, lds2, stream>>>(z_number, atom_embed, W_cat,
                                            b_cat, aggr, out, N);
}
